// KPCNDenoiser_2886218023189
// MI455X (gfx1250) — matrix operand order repaired and checked by reading, not compiled, not run on hardware
//
#include <hip/hip_runtime.h>
#include <hip/hip_bf16.h>
#include <math.h>
#include <stdint.h>

// ---------------------------------------------------------------------------
// Types for CDNA5 WMMA (wave32, 16x16x32 bf16 -> f32) and TDM descriptors
// ---------------------------------------------------------------------------
typedef __bf16          v16bf __attribute__((ext_vector_type(16)));
typedef unsigned short  v16us __attribute__((ext_vector_type(16)));
typedef float           v8f   __attribute__((ext_vector_type(8)));
typedef unsigned int    u32x4 __attribute__((ext_vector_type(4)));
typedef int             i32x4 __attribute__((ext_vector_type(4)));
typedef int             i32x8 __attribute__((ext_vector_type(8)));

#if defined(__has_builtin)
#  if __has_builtin(__builtin_amdgcn_tensor_load_to_lds) && \
      __has_builtin(__builtin_amdgcn_s_wait_tensorcnt)
#    define HAS_TDM 1
#  endif
#endif
#ifndef HAS_TDM
#  define HAS_TDM 0
#endif

// Definitive probe: fail the DEVICE pass only if the TDM builtin is missing,
// so we know from the compile result whether the async-tensor path lowered.
#if defined(__gfx1250__) && !HAS_TDM
#  error "gfx1250 device pass without __builtin_amdgcn_tensor_load_to_lds"
#endif

__device__ __forceinline__ unsigned short f2bf(float f) {
    unsigned int u = __builtin_bit_cast(unsigned int, f);
    unsigned int r = u + 0x7FFFu + ((u >> 16) & 1u);   // round-to-nearest-even
    return (unsigned short)(r >> 16);
}

#if HAS_TDM
// Issue a TDM 2D tile load: 32 rows x 64 cols of fp32 from a [rows x Ktot]
// row-major matrix starting at gsrc, into LDS at ldsOff with 1-dword padding
// after every 64 dwords (row stride in LDS becomes 65 dwords).
// remK/remRows = remaining tensor extent from gsrc for OOB zero-fill.
__device__ __forceinline__ void tdm_load_wtile(const float* gsrc, unsigned ldsOff,
                                               int remK, int remRows, int Ktot) {
    unsigned long long ga = (unsigned long long)(uintptr_t)gsrc;
    u32x4 g0;
    g0[0] = 1u;                                        // count=1, user descriptor
    g0[1] = ldsOff;                                    // lds_addr (bytes)
    g0[2] = (unsigned)ga;                              // global_addr[31:0]
    g0[3] = (unsigned)((ga >> 32) & 0x01FFFFFFull)     // global_addr[56:32]
          | (2u << 30);                                // type = 2 ("image")
    i32x8 g1;
    g1[0] = (int)(0x00020000u                          // data_size = 4 bytes
          | (1u << 20)                                 // pad_enable
          | (5u << 22)                                 // pad_interval: 64 dwords
          | (0u << 25));                               // pad_amount: 1 dword
    g1[1] = (int)(((unsigned)remK & 0xFFFFu) << 16);   // tensor_dim0[15:0] @63:48
    g1[2] = (int)(((unsigned)remK >> 16)               // tensor_dim0[31:16]
          | (((unsigned)remRows & 0xFFFFu) << 16));    // tensor_dim1[15:0]
    g1[3] = (int)(((unsigned)remRows >> 16)            // tensor_dim1[31:16]
          | (64u << 16));                              // tile_dim0 = 64
    g1[4] = 32;                                        // tile_dim1 = 32
    g1[5] = Ktot;                                      // tensor_dim0_stride[31:0]
    g1[6] = 0;
    g1[7] = 0;
    i32x4 z4 = {0, 0, 0, 0};
#if defined(__clang_major__) && __clang_major__ >= 23
    i32x8 z8 = {0, 0, 0, 0, 0, 0, 0, 0};
    __builtin_amdgcn_tensor_load_to_lds(g0, g1, z4, z4, z8, 0);
#else
    __builtin_amdgcn_tensor_load_to_lds(g0, g1, z4, z4, 0);
#endif
}
#endif

// ---------------------------------------------------------------------------
// Implicit-GEMM conv (ks = 1 or 3, pad = ks/2), NCHW fp32 in/out.
// 256-thread block (8 waves) computes a 128-pixel x 32-outchannel macro-tile.
//   wm = wave&3 selects a 32-pixel M strip (two 16-row subtiles, register
//   blocked), wn = wave>>2 selects the 16-channel N subtile.
// Per 64-wide K chunk each wave issues 4 WMMAs; the B fragment is built once
// per sub-chunk and reused across both M subtiles.
// A (128x64, bf16 im2col) staged cooperatively in LDS, shared by N-columns.
// B (32x64 weights, fp32) DMA'd by the Tensor Data Mover into LDS with
// 1-dword/row padding (65-dword stride -> conflict-free reads), overlapped
// with the VALU im2col staging, then s_wait_tensorcnt + barrier.
// ---------------------------------------------------------------------------
__global__ void __launch_bounds__(256)
conv_wmma(const float* __restrict__ in, const float* __restrict__ w,
          const float* __restrict__ bias, float* __restrict__ out,
          int N, int Cin, int CinTot, int Cout, int H, int W,
          int ks, int relu)
{
    const int tid    = threadIdx.x;
    const int lane   = tid & 31;
    const int wave   = tid >> 5;
    const int wm     = wave & 3;                // 32-pixel M strip 0..3
    const int wn     = wave >> 2;               // N subtile 0..1
    const int mbase  = blockIdx.x * 128;        // pixel tile base (flat n*H*W)
    const int cobase = blockIdx.y * 32;         // out-channel tile base
    const int HW     = H * W;
    const int M      = N * HW;
    const int KK     = ks * ks;
    const int Ktot   = Cin * KK;
    const int pad    = ks >> 1;

    __shared__ unsigned short As[128][68];      // 128 px x 64 K (bf16, +pad)
    __shared__ float          Bs[32 * 65 + 64]; // 32 rows x 64 K (fp32, str 65)

    v8f acc0 = {}, acc1 = {};
    const int half8 = (lane >> 4) << 3;         // 0 for lanes 0-15, 8 for 16-31
    const int nIdx  = lane & 15;
    const int aRow0 = wm * 32 + (lane & 15);    // first M subtile row
    const int aRow1 = aRow0 + 16;               // second M subtile row
    const int bRow  = wn * 16 + (lane & 15);    // B row (channel within tile)

    for (int kb = 0; kb < Ktot; kb += 64) {
        // ---- B tile: TDM DMA (wave 0 issues; OOB rows/cols zero-fill) ----
#if HAS_TDM
        if (wave == 0) {
            tdm_load_wtile(w + (long)cobase * Ktot + kb,
                           (unsigned)(unsigned long long)(uintptr_t)&Bs[0],
                           Ktot - kb, Cout - cobase, Ktot);
        }
#else
        for (int it = 0; it < 8; ++it) {
            int idx = it * 256 + tid;
            int row = idx >> 6, k = idx & 63;
            int co = cobase + row, kidx = kb + k;
            float v = (co < Cout && kidx < Ktot) ? w[(long)co * Ktot + kidx] : 0.0f;
            Bs[row * 65 + k] = v;
        }
#endif
        // ---- A tile: im2col 128 pixels x 64 K, bf16 (overlaps TDM) ----
        for (int it = 0; it < 32; ++it) {
            int idx = it * 256 + tid;           // 0..8191, uniform trip count
            int ml  = idx >> 6;
            int kk  = idx & 63;
            int kidx = kb + kk;
            float v = 0.0f;
            int m = mbase + ml;
            if (kidx < Ktot && m < M) {
                int c  = kidx / KK;
                int r  = kidx - c * KK;
                int ry = r / ks;
                int dy = ry - pad;
                int dx = (r - ry * ks) - pad;
                int n  = m / HW;
                int p  = m - n * HW;
                int py = p / W;
                int y  = py + dy;
                int x  = (p - py * W) + dx;
                if ((unsigned)y < (unsigned)H && (unsigned)x < (unsigned)W)
                    v = in[((long)n * CinTot + c) * HW + (long)y * W + x];
            }
            As[ml][kk] = f2bf(v);
        }
#if HAS_TDM
        if (wave == 0) __builtin_amdgcn_s_wait_tensorcnt(0);
#endif
        __syncthreads();

        // ---- 4 WMMAs: 2 K-sub-chunks x 2 M-subtiles (B frag reused) ----
        #pragma unroll
        for (int s = 0; s < 2; ++s) {
            v16us bu;
            #pragma unroll
            for (int e = 0; e < 16; ++e) {
                // B 32x16 bf16: K = (e&7) + (e>>3)*16 + (lane<16 ? 0 : 8)
                int kbk = s * 32 + (e & 7) + ((e >> 3) << 4) + half8;
                bu[e] = f2bf(Bs[bRow * 65 + kbk]);
            }
            v16bf bv = __builtin_bit_cast(v16bf, bu);

            v16us au0, au1;
            #pragma unroll
            for (int e = 0; e < 16; ++e) {
                // A 16x32 bf16: K = (e&7) + (e>>3)*16 + (lane<16 ? 0 : 8)
                int ka = s * 32 + (e & 7) + ((e >> 3) << 4) + half8;
                au0[e] = As[aRow0][ka];
                au1[e] = As[aRow1][ka];
            }
            v16bf av0 = __builtin_bit_cast(v16bf, au0);
            v16bf av1 = __builtin_bit_cast(v16bf, au1);
            acc0 = __builtin_amdgcn_wmma_f32_16x16x32_bf16(
                       false, av0, false, bv, (short)0, acc0, false, false);
            acc1 = __builtin_amdgcn_wmma_f32_16x16x32_bf16(
                       false, av1, false, bv, (short)0, acc1, false, false);
        }
        __syncthreads();
    }

    // ---- epilogue: bias (+ReLU), scatter both D tiles ----
    const int co = cobase + wn * 16 + nIdx;
    if (co < Cout) {
        float bv = bias[co];
        #pragma unroll
        for (int j = 0; j < 8; ++j) {
            int m0 = mbase + wm * 32 + j + half8;   // D: M = j + (lane<16?0:8)
            if (m0 < M) {
                int n = m0 / HW, p = m0 - n * HW;
                float v = acc0[j] + bv;
                if (relu) v = v > 0.0f ? v : 0.0f;
                out[((long)n * Cout + co) * HW + p] = v;
            }
            int m1 = m0 + 16;
            if (m1 < M) {
                int n = m1 / HW, p = m1 - n * HW;
                float v = acc1[j] + bv;
                if (relu) v = v > 0.0f ? v : 0.0f;
                out[((long)n * Cout + co) * HW + p] = v;
            }
        }
    }
}

// ---------------------------------------------------------------------------
// Fused kernel-prediction head: conv1x1(e2) + softmax(441) + 21x21 KPN filter.
// Streaming (online) softmax keeps everything in registers; hidden vector
// tile lives in LDS. Avoids materializing the 441-channel tensors
// (saves >3 GB of HBM round-trips vs the naive pipeline).
// ---------------------------------------------------------------------------
#define KRAD  10
#define KDIAM 21
#define KSUP  441

__global__ void __launch_bounds__(64)
kpn_fused(const float* __restrict__ color,    // [N,3,H,W]
          const float* __restrict__ hidden,   // [N,Ch,H,W]
          const float* __restrict__ w2,       // [441, Ch]
          const float* __restrict__ b2,       // [441]
          float* __restrict__ out,            // [N,3,H,W]
          int N, int Ch, int H, int W)
{
    __shared__ float hs[192 * 64];            // Ch <= 192 hidden, 64 pixels
    const int t  = threadIdx.x;
    const int HW = H * W;
    const long M = (long)N * HW;
    long pix = (long)blockIdx.x * 64 + t;
    bool active = pix < M;
    long pp = active ? pix : (M - 1);
    int n = (int)(pp / HW);
    int p = (int)(pp - (long)n * HW);
    int y = p / W;
    int x = p - y * W;

    for (int c = 0; c < Ch; ++c)
        hs[c * 64 + t] = hidden[((long)n * Ch + c) * HW + p];
    __syncthreads();

    float mx = -3.4e38f, s = 0.0f, a0 = 0.0f, a1 = 0.0f, a2 = 0.0f;
    const float* cb = color + (long)n * 3 * HW;

    for (int i = 0; i < KSUP; ++i) {
        const float* wr = w2 + (long)i * Ch;
        if (i + 1 < KSUP)
            __builtin_prefetch(wr + Ch, 0, 0);        // global_prefetch_b8
        float logit = b2[i];
        #pragma unroll 4
        for (int c = 0; c < Ch; ++c)
            logit = fmaf(hs[c * 64 + t], wr[c], logit);

        float mn = logit > mx ? logit : mx;
        float sc = __expf(mx - mn);
        float pw = __expf(logit - mn);

        int iy = i / KDIAM;
        int dy = iy - KRAD;
        int dx = (i - iy * KDIAM) - KRAD;
        int yy = y + dy, xx = x + dx;
        float c0 = 0.0f, c1 = 0.0f, c2 = 0.0f;
        if ((unsigned)yy < (unsigned)H && (unsigned)xx < (unsigned)W) {
            long q = (long)yy * W + xx;
            c0 = cb[q]; c1 = cb[HW + q]; c2 = cb[2 * HW + q];
        }
        s  = s  * sc + pw;
        a0 = a0 * sc + pw * c0;
        a1 = a1 * sc + pw * c1;
        a2 = a2 * sc + pw * c2;
        mx = mn;
    }

    if (active) {
        float inv = 1.0f / s;
        long ob = (long)n * 3 * HW + p;
        out[ob]          = a0 * inv;
        out[ob + HW]     = a1 * inv;
        out[ob + 2 * HW] = a2 * inv;
    }
}

// ---------------------------------------------------------------------------
// Glue kernels (bandwidth-bound elementwise / pooling / resampling)
// ---------------------------------------------------------------------------
__global__ void k_slice_fc(const float* __restrict__ x, float* __restrict__ fc,
                           long total, int HW3) {           // fc = x[:,0:3]
    long i = (long)blockIdx.x * blockDim.x + threadIdx.x;
    if (i >= total) return;
    long n = i / HW3, rem = i - n * HW3;
    fc[i] = x[n * (long)HW3 * 3 + rem];                     // 9HW = 3*HW3
}

__global__ void k_build_tin(const float* __restrict__ x, float* __restrict__ tin,
                            long total, int HW) {           // [N,9,H,W]
    long i = (long)blockIdx.x * blockDim.x + threadIdx.x;
    if (i >= total) return;
    int c = (int)((i / HW) % 9);
    float v = x[i];
    tin[i] = (c < 3) ? log1pf(v) : v;
}

__global__ void k_pool2(const float* __restrict__ in, float* __restrict__ out,
                        long total, int Ho, int Wo, int isMax) {
    long i = (long)blockIdx.x * blockDim.x + threadIdx.x;
    if (i >= total) return;
    long nc = i / ((long)Ho * Wo);
    long r  = i - nc * (long)Ho * Wo;
    int y = (int)(r / Wo), x = (int)(r - (long)y * Wo);
    const float* b = in + nc * (long)(2 * Ho) * (2 * Wo);
    long q = (long)(2 * y) * (2 * Wo) + 2 * x;
    float a = b[q], bb = b[q + 1], c = b[q + 2 * Wo], d = b[q + 2 * Wo + 1];
    if (isMax) {
        float m1 = a > bb ? a : bb, m2 = c > d ? c : d;
        out[i] = m1 > m2 ? m1 : m2;
    } else {
        out[i] = 0.25f * (a + bb + c + d);
    }
}

__global__ void k_upcat(const float* __restrict__ a, const float* __restrict__ b,
                        float* __restrict__ out, long total,
                        int N, int C1, int C2, int H, int W) {
    long i = (long)blockIdx.x * blockDim.x + threadIdx.x;
    if (i >= total) return;
    int C = C1 + C2;
    long HW = (long)H * W;
    long n = i / ((long)C * HW);
    long r = i - n * (long)C * HW;
    int c = (int)(r / HW);
    long pr = r - (long)c * HW;
    int y = (int)(pr / W), x = (int)(pr - (long)y * W);
    float v;
    if (c < C1) {
        int Hc = H >> 1, Wc = W >> 1;
        v = a[((n * C1 + c) * Hc + (y >> 1)) * Wc + (x >> 1)];
    } else {
        v = b[((n * C2 + (c - C1)) * (long)H + y) * W + x];
    }
    out[i] = v;
}

__global__ void k_residual(const float* __restrict__ prev, const float* __restrict__ f,
                           float* __restrict__ r, long total, int Hc, int Wc) {
    long i = (long)blockIdx.x * blockDim.x + threadIdx.x;
    if (i >= total) return;
    long nc = i / ((long)Hc * Wc);
    long rr = i - nc * (long)Hc * Wc;
    int y = (int)(rr / Wc), x = (int)(rr - (long)y * Wc);
    const float* fb = f + nc * (long)(2 * Hc) * (2 * Wc);
    long q = (long)(2 * y) * (2 * Wc) + 2 * x;
    float avg = 0.25f * (fb[q] + fb[q + 1] + fb[q + 2 * Wc] + fb[q + 2 * Wc + 1]);
    r[i] = prev[i] - avg;
}

__device__ __forceinline__ int iclampi(int v, int lo, int hi) {
    return v < lo ? lo : (v > hi ? hi : v);
}

__global__ void k_combine(const float* __restrict__ f, const float* __restrict__ cw,
                          long cwStride, const float* __restrict__ r,
                          float* __restrict__ out, long total,
                          int H, int W, int tmap) {
    long i = (long)blockIdx.x * blockDim.x + threadIdx.x;
    if (i >= total) return;
    long HW = (long)H * W;
    long nc = i / HW;
    long n  = nc / 3;
    long pr = i - nc * HW;
    int y = (int)(pr / W), x = (int)(pr - (long)y * W);
    int Hc = H >> 1, Wc = W >> 1;
    // half-pixel bilinear, edge clamp (matches jax.image.resize 2x)
    float sy = (y + 0.5f) * 0.5f - 0.5f;
    float sx = (x + 0.5f) * 0.5f - 0.5f;
    int y0 = (int)floorf(sy), x0 = (int)floorf(sx);
    float wy = sy - y0, wx = sx - x0;
    int y0c = iclampi(y0, 0, Hc - 1), y1c = iclampi(y0 + 1, 0, Hc - 1);
    int x0c = iclampi(x0, 0, Wc - 1), x1c = iclampi(x0 + 1, 0, Wc - 1);
    const float* rb = r + nc * (long)Hc * Wc;
    float v00 = rb[(long)y0c * Wc + x0c], v01 = rb[(long)y0c * Wc + x1c];
    float v10 = rb[(long)y1c * Wc + x0c], v11 = rb[(long)y1c * Wc + x1c];
    float samp = v00 * (1 - wy) * (1 - wx) + v01 * (1 - wy) * wx
               + v10 * wy * (1 - wx)       + v11 * wy * wx;
    float cwv = cw[n * cwStride + pr];
    float v = f[i] + cwv * samp;
    out[i] = tmap ? log1pf(v) : v;
}

// ---------------------------------------------------------------------------
// Host orchestration
// ---------------------------------------------------------------------------
static inline int cdiv(long a, long b) { return (int)((a + b - 1) / b); }

static void conv(hipStream_t s, const float* in, const float* w, const float* b,
                 float* out, int N, int Cin, int CinTot, int Cout,
                 int H, int W, int ks, int relu) {
    dim3 g(cdiv((long)N * H * W, 128), cdiv(Cout, 32));
    conv_wmma<<<g, dim3(256), 0, s>>>(in, w, b, out, N, Cin, CinTot, Cout, H, W, ks, relu);
}

#define EW(kern, total, ...) do {                                            \
    long _t = (total);                                                       \
    kern<<<cdiv(_t, 256), 256, 0, stream>>>(__VA_ARGS__);                    \
} while (0)

extern "C" void kernel_launch(void* const* d_in, const int* in_sizes, int n_in,
                              void* d_out, int out_size, void* d_ws, size_t ws_size,
                              hipStream_t stream) {
    (void)in_sizes; (void)n_in; (void)out_size; (void)ws_size;
    const int N = 2, F = 384, Hh = 192, Q = 96, E = 48, T = 24;
    const long F2 = (long)F * F, H2 = (long)Hh * Hh, Q2 = (long)Q * Q,
               E2 = (long)E * E, T2 = (long)T * T;

    const float* x = (const float*)d_in[0];
    auto W_ = [&](int i) { return (const float*)d_in[i]; };
    // param indices (dict insertion order after 'x')
    enum { enc0a_w = 1, enc0a_b, enc0b_w, enc0b_b, enc1_w, enc1_b, enc2_w, enc2_b,
           enc3_w, enc3_b, enc4_w, enc4_b, dec5_w, dec5_b, dec4a_w, dec4a_b,
           dec4b_w, dec4b_b, dec3a_w, dec3a_b, dec3b_w, dec3b_b, dec2a_w, dec2a_b,
           dec2b_w, dec2b_b, dec1a_w, dec1a_b, dec1b_w, dec1b_b, dec0_w, dec0_b,
           k2e1_w, k2e1_b, k2e2_w, k2e2_b, k1e1_w, k1e1_b, k1e2_w, k1e2_b,
           k0e1_w, k0e1_b, k0e2_w, k0e2_b };

    // bump allocator in workspace
    size_t off = 0;
    auto alloc = [&](long nfloats) {
        float* p = (float*)((char*)d_ws + off);
        off += ((size_t)nfloats * 4 + 255) & ~(size_t)255;
        return p;
    };

    float* fc   = alloc(N * 3 * F2);
    float* hc   = alloc(N * 3 * H2);
    float* qc   = alloc(N * 3 * Q2);
    float* tin  = alloc(N * 9 * F2);
    float* e0a  = alloc(N * 32 * F2);
    float* e0b  = alloc(N * 32 * F2);
    float* p1   = alloc(N * 32 * H2);
    float* e1   = alloc(N * 48 * H2);
    float* p2   = alloc(N * 48 * Q2);
    float* e2   = alloc(N * 64 * Q2);
    float* p3   = alloc(N * 64 * E2);
    float* e3   = alloc(N * 80 * E2);
    float* p4   = alloc(N * 80 * T2);
    float* e4   = alloc(N * 96 * T2);
    float* d5   = alloc(N * 96 * T2);
    float* cat4 = alloc(N * 160 * E2);
    float* d4a  = alloc(N * 112 * E2);
    float* d4b  = alloc(N * 112 * E2);
    float* cat3 = alloc(N * 160 * Q2);
    float* d3a  = alloc(N * 96 * Q2);
    float* qk   = alloc(N * 96 * Q2);
    float* cat2 = alloc(N * 128 * H2);
    float* d2a  = alloc(N * 64 * H2);
    float* hk   = alloc(N * 64 * H2);
    float* cat1 = alloc(N * 73 * F2);
    float* d1a  = alloc(N * 64 * F2);
    float* d1b  = alloc(N * 32 * F2);
    float* fk   = alloc(N * 32 * F2);
    float* h2   = alloc(N * 192 * Q2);
    float* f2v  = alloc(N * 3 * Q2);
    float* h1   = alloc(N * 126 * H2);
    float* f1v  = alloc(N * 3 * H2);
    float* r2   = alloc(N * 3 * Q2);
    float* cb1  = alloc(N * 3 * H2);
    float* h0   = alloc(N * 93 * F2);
    float* f0v  = alloc(N * 3 * F2);
    float* r1   = alloc(N * 3 * H2);
    float* outp = (float*)d_out;

    // ---- input prep ----
    EW(k_slice_fc, N * 3 * F2, x, fc, N * 3 * F2, (int)(3 * F2));
    EW(k_pool2, N * 3 * H2, fc, hc, N * 3 * H2, Hh, Hh, 0);
    EW(k_pool2, N * 3 * Q2, hc, qc, N * 3 * Q2, Q, Q, 0);
    EW(k_build_tin, N * 9 * F2, x, tin, N * 9 * F2, (int)F2);

    // ---- encoder ----
    conv(stream, tin, W_(enc0a_w), W_(enc0a_b), e0a, N, 9, 9, 32, F, F, 3, 1);
    conv(stream, e0a, W_(enc0b_w), W_(enc0b_b), e0b, N, 32, 32, 32, F, F, 3, 1);
    EW(k_pool2, N * 32 * H2, e0b, p1, N * 32 * H2, Hh, Hh, 1);
    conv(stream, p1, W_(enc1_w), W_(enc1_b), e1, N, 32, 32, 48, Hh, Hh, 3, 1);
    EW(k_pool2, N * 48 * Q2, e1, p2, N * 48 * Q2, Q, Q, 1);
    conv(stream, p2, W_(enc2_w), W_(enc2_b), e2, N, 48, 48, 64, Q, Q, 3, 1);
    EW(k_pool2, N * 64 * E2, e2, p3, N * 64 * E2, E, E, 1);
    conv(stream, p3, W_(enc3_w), W_(enc3_b), e3, N, 64, 64, 80, E, E, 3, 1);
    EW(k_pool2, N * 80 * T2, e3, p4, N * 80 * T2, T, T, 1);
    conv(stream, p4, W_(enc4_w), W_(enc4_b), e4, N, 80, 80, 96, T, T, 3, 1);
    conv(stream, e4, W_(dec5_w), W_(dec5_b), d5, N, 96, 96, 96, T, T, 3, 1);

    // ---- decoder ----
    EW(k_upcat, N * 160 * E2, d5, p3, cat4, N * 160 * E2, N, 96, 64, E, E);
    conv(stream, cat4, W_(dec4a_w), W_(dec4a_b), d4a, N, 160, 160, 112, E, E, 3, 1);
    conv(stream, d4a, W_(dec4b_w), W_(dec4b_b), d4b, N, 112, 112, 112, E, E, 3, 1);
    EW(k_upcat, N * 160 * Q2, d4b, p2, cat3, N * 160 * Q2, N, 112, 48, Q, Q);
    conv(stream, cat3, W_(dec3a_w), W_(dec3a_b), d3a, N, 160, 160, 96, Q, Q, 3, 1);
    conv(stream, d3a, W_(dec3b_w), W_(dec3b_b), qk, N, 96, 96, 96, Q, Q, 3, 1);
    EW(k_upcat, N * 128 * H2, qk, p1, cat2, N * 128 * H2, N, 96, 32, Hh, Hh);
    conv(stream, cat2, W_(dec2a_w), W_(dec2a_b), d2a, N, 128, 128, 64, Hh, Hh, 3, 1);
    conv(stream, d2a, W_(dec2b_w), W_(dec2b_b), hk, N, 64, 64, 64, Hh, Hh, 3, 1);
    EW(k_upcat, N * 73 * F2, hk, tin, cat1, N * 73 * F2, N, 64, 9, F, F);
    conv(stream, cat1, W_(dec1a_w), W_(dec1a_b), d1a, N, 73, 73, 64, F, F, 3, 1);
    conv(stream, d1a, W_(dec1b_w), W_(dec1b_b), d1b, N, 64, 64, 32, F, F, 3, 1);
    conv(stream, d1b, W_(dec0_w), W_(dec0_b), fk, N, 32, 32, 32, F, F, 3, 0);

    // ---- kernel-prediction heads (fused conv1x1 + softmax + 21x21 filter) ----
    conv(stream, qk, W_(k2e1_w), W_(k2e1_b), h2, N, 96, 96, 192, Q, Q, 1, 1);
    kpn_fused<<<cdiv((long)N * Q2, 64), 64, 0, stream>>>(
        qc, h2, W_(k2e2_w), W_(k2e2_b), f2v, N, 192, Q, Q);

    conv(stream, hk, W_(k1e1_w), W_(k1e1_b), h1, N, 63, 64, 126, Hh, Hh, 1, 1);
    kpn_fused<<<cdiv((long)N * H2, 64), 64, 0, stream>>>(
        hc, h1, W_(k1e2_w), W_(k1e2_b), f1v, N, 126, Hh, Hh);

    // filtered = f1 + cw1 * bilin(f2 - avgpool2(f1))
    EW(k_residual, N * 3 * Q2, f2v, f1v, r2, N * 3 * Q2, Q, Q);
    EW(k_combine, N * 3 * H2, f1v, hk + 63 * H2, 64 * H2, r2, cb1,
       N * 3 * H2, Hh, Hh, 0);

    conv(stream, fk, W_(k0e1_w), W_(k0e1_b), h0, N, 31, 32, 93, F, F, 1, 1);
    kpn_fused<<<cdiv((long)N * F2, 64), 64, 0, stream>>>(
        fc, h0, W_(k0e2_w), W_(k0e2_b), f0v, N, 93, F, F);

    // final = log1p(f0 + cw0 * bilin(cb1 - avgpool2(f0)))
    EW(k_residual, N * 3 * H2, cb1, f0v, r1, N * 3 * H2, Hh, Hh);
    EW(k_combine, N * 3 * F2, f0v, fk + 31 * F2, 32 * F2, r1, outp,
       N * 3 * F2, F, F, 1);
}
